// Block_SpeGroup_27685359190799
// MI455X (gfx1250) — compile-verified, matrix operand order checked by reading, not run on hardware
//
#include <hip/hip_runtime.h>
#include <hip/hip_bf16.h>
#include <stdint.h>

// Problem dimensions (compile-time, from the reference)
#define BB   32
#define HH_  32
#define WW_  32
#define CDIM 128
#define KDIR 4
#define NST  16
#define RLOW 2
#define DSV  32
#define LSEQ 1024          // d4 * H = 32*32
#define PIX  (BB*HH_*WW_)  // 32768

typedef __attribute__((ext_vector_type(16))) __bf16 v16bf;
typedef __attribute__((ext_vector_type(8)))  float  v8f;
typedef __attribute__((ext_vector_type(4)))  unsigned u32x4;
typedef __attribute__((ext_vector_type(8)))  unsigned u32x8;

__device__ __forceinline__ __bf16 f2bf(float f) { return (__bf16)f; }  // native v_cvt RTNE
__device__ __forceinline__ float sigmoidf_(float x){ return 1.0f/(1.0f+__expf(-x)); }
__device__ __forceinline__ float siluf_(float x){ return x*sigmoidf_(x); }
__device__ __forceinline__ float softplusf_(float x){ return (x>20.0f)?x:log1pf(__expf(x)); }

// ---------------------------------------------------------------------------
// TDM: 2-D tile load (global -> LDS) via tensor_load_to_lds with an inline
// D# descriptor (group0: 4 SGPRs, group1: 8 SGPRs). data_size = 4 bytes.
//   tile: tile_d0 (contiguous elems) x tile_d1 (rows), row stride stride_d0.
// Issued per-wave (EXEC ignored); tracked with TENSORcnt.
// ---------------------------------------------------------------------------
__device__ __forceinline__ void tdm_load_2d(const void* gsrc, void* lds_dst,
                                            unsigned tile_d0, unsigned tile_d1,
                                            unsigned tensor_d0, unsigned tensor_d1,
                                            unsigned long long stride_d0) {
  unsigned long long ga   = (unsigned long long)(uintptr_t)gsrc;
  unsigned           ldso = (unsigned)(uintptr_t)lds_dst;

  // group0 (128b): [1:0]=count=1, [63:32]=lds_addr, [120:64]=global_addr, [127:126]=type=2
  unsigned long long q0 = 1ull | ((unsigned long long)ldso << 32);
  unsigned long long q1 = (ga & 0x01ffffffffffffffull) | (2ull << 62);
  // group1 (256b): [17:16]=data_size=2 (4B), [79:48]=tensor_dim0, [111:80]=tensor_dim1,
  //                [127:112]=tile_dim0, [143:128]=tile_dim1, [207:160]=tensor_dim0_stride
  unsigned long long q2 = (2ull << 16) |
                          (((unsigned long long)(tensor_d0 & 0xffffu)) << 48);
  unsigned long long q3 = ((unsigned long long)(tensor_d0 >> 16)) |
                          (((unsigned long long)tensor_d1) << 16) |
                          (((unsigned long long)(tile_d0 & 0xffffu)) << 48);
  unsigned long long q4 = ((unsigned long long)(tile_d1 & 0xffffu)) |
                          ((stride_d0 & 0xffffffffull) << 32);
  unsigned long long q5 = (stride_d0 >> 32) & 0xffffull;

  u32x4 g0 = { (unsigned)q0, (unsigned)(q0 >> 32), (unsigned)q1, (unsigned)(q1 >> 32) };
  u32x8 g1 = { (unsigned)q2, (unsigned)(q2 >> 32), (unsigned)q3, (unsigned)(q3 >> 32),
               (unsigned)q4, (unsigned)(q4 >> 32), (unsigned)q5, (unsigned)(q5 >> 32) };
  asm volatile("tensor_load_to_lds %0, %1" :: "s"(g0), "s"(g1) : "memory");
}

// ---------------------------------------------------------------------------
// WMMA bf16 GEMM:  C[M,N] = A[M,K] * Bt[N,K]^T   (f32 in/out, bf16 matrix math)
// One wave computes a 16x64 strip (4 accumulators); K fully unrolled.
// A fragment (16x32, 16-bit): lane m=lane&15; lanes<16 carry K {0..7,16..23},
// lanes>=16 carry K {8..15,24..31}. B fragment mirrors with n=lane&15.
// ---------------------------------------------------------------------------
template<int KD, int NCOLS>
__global__ void wmma_gemm_bt(const float* __restrict__ A,
                             const float* __restrict__ Bt,
                             float* __restrict__ C, int M) {
  int wave = (int)((blockIdx.x * blockDim.x + threadIdx.x) >> 5);
  int lane = (int)(threadIdx.x & 31);
  const int tilesN4 = NCOLS >> 6;                 // 64-column strips
  int totalWaves = (M >> 4) * tilesN4;
  if (wave >= totalWaves) return;
  int tm  = wave / tilesN4;
  int tn4 = wave % tilesN4;

  int mrow = (tm << 4) + (lane & 15);
  int kb   = (lane < 16) ? 0 : 8;
  const float* arow = A + (size_t)mrow * KD;

  v8f acc[4] = {};
#pragma unroll
  for (int kk = 0; kk < KD; kk += 32) {
    const float4 a0 = *reinterpret_cast<const float4*>(arow + kk + kb);
    const float4 a1 = *reinterpret_cast<const float4*>(arow + kk + kb + 4);
    const float4 a2 = *reinterpret_cast<const float4*>(arow + kk + kb + 16);
    const float4 a3 = *reinterpret_cast<const float4*>(arow + kk + kb + 20);
    v16bf af;
    af[0]=f2bf(a0.x);  af[1]=f2bf(a0.y);  af[2]=f2bf(a0.z);  af[3]=f2bf(a0.w);
    af[4]=f2bf(a1.x);  af[5]=f2bf(a1.y);  af[6]=f2bf(a1.z);  af[7]=f2bf(a1.w);
    af[8]=f2bf(a2.x);  af[9]=f2bf(a2.y);  af[10]=f2bf(a2.z); af[11]=f2bf(a2.w);
    af[12]=f2bf(a3.x); af[13]=f2bf(a3.y); af[14]=f2bf(a3.z); af[15]=f2bf(a3.w);
#pragma unroll
    for (int tI = 0; tI < 4; ++tI) {
      const float* brow = Bt + (size_t)((tn4 << 6) + (tI << 4) + (lane & 15)) * KD;
      const float4 b0 = *reinterpret_cast<const float4*>(brow + kk + kb);
      const float4 b1 = *reinterpret_cast<const float4*>(brow + kk + kb + 4);
      const float4 b2 = *reinterpret_cast<const float4*>(brow + kk + kb + 16);
      const float4 b3 = *reinterpret_cast<const float4*>(brow + kk + kb + 20);
      v16bf bf;
      bf[0]=f2bf(b0.x);  bf[1]=f2bf(b0.y);  bf[2]=f2bf(b0.z);  bf[3]=f2bf(b0.w);
      bf[4]=f2bf(b1.x);  bf[5]=f2bf(b1.y);  bf[6]=f2bf(b1.z);  bf[7]=f2bf(b1.w);
      bf[8]=f2bf(b2.x);  bf[9]=f2bf(b2.y);  bf[10]=f2bf(b2.z); bf[11]=f2bf(b2.w);
      bf[12]=f2bf(b3.x); bf[13]=f2bf(b3.y); bf[14]=f2bf(b3.z); bf[15]=f2bf(b3.w);
      acc[tI] = __builtin_amdgcn_wmma_f32_16x16x32_bf16(
          false, af, false, bf, (short)0, acc[tI], false, false);
    }
  }

  int mb = (tm << 4) + ((lane < 16) ? 0 : 8);
#pragma unroll
  for (int tI = 0; tI < 4; ++tI) {
    int nc = (tn4 << 6) + (tI << 4) + (lane & 15);
#pragma unroll
    for (int r = 0; r < 8; ++r)
      C[(size_t)(mb + r) * NCOLS + nc] = acc[tI][r];
  }
}

// ---------------------------------------------------------------------------
// k2: silu(z), conv scale/bias + silu, squeeze sums, cross-scan scatter to xs
// ---------------------------------------------------------------------------
__global__ void k2_pre(const float* __restrict__ xz, const float* __restrict__ cw,
                       const float* __restrict__ cb, float* __restrict__ zs,
                       float* __restrict__ xs, float* __restrict__ zzsum) {
  int idx = (int)(blockIdx.x * blockDim.x + threadIdx.x);
  if (idx >= PIX * CDIM) return;
  int c   = idx & (CDIM - 1);
  int pix = idx >> 7;
  int ww  = pix & 31;
  int b   = pix >> 10;

  float xx = xz[(size_t)pix * 256 + c];
  float zr = xz[(size_t)pix * 256 + 128 + c];
  zs[idx]  = siluf_(zr);

  float v = siluf_(xx * cw[c] + cb[c]);
  atomicAdd(&zzsum[b * CDIM + c], v);

  int hh = (pix >> 5) & 31;
  int dd = c >> 2, g = c & 3;
  int l0 = dd * 32 + ww;
  int l1 = ww * 32 + dd;
  int k, l;
  if (g == 0)      { k = 0; l = l0; }
  else if (g == 1) { k = 1; l = l1; }
  else if (g == 2) { k = 2; l = 1023 - l0; }
  else             { k = 3; l = 1023 - l1; }
  xs[(((size_t)b * KDIR + k) * DSV + hh) * LSEQ + l] = v;
}

// ---------------------------------------------------------------------------
// k3: squeeze-excite  (one block per batch, 128 threads)
// ---------------------------------------------------------------------------
__global__ void k3_se(const float* __restrict__ zzsum,
                      const float* __restrict__ fc1_w, const float* __restrict__ fc1_b,
                      const float* __restrict__ fc2_w, const float* __restrict__ fc2_b,
                      float* __restrict__ f2out) {
  int b = blockIdx.x;
  int d = threadIdx.x;
  __shared__ float szz[CDIM];
  __shared__ float sf1[4];
  szz[d] = zzsum[b * CDIM + d] * (1.0f / 1024.0f);
  __syncthreads();
  if (d < 4) {
    float s = fc1_b[d];
    for (int i = 0; i < CDIM; ++i) s += szz[i] * fc1_w[d * CDIM + i];
    sf1[d] = fmaxf(s, 0.0f);
  }
  __syncthreads();
  float s = fc2_b[d];
#pragma unroll
  for (int j = 0; j < 4; ++j) s += sf1[j] * fc2_w[d * 4 + j];
  f2out[b * CDIM + d] = sigmoidf_(s);
}

// ---------------------------------------------------------------------------
// k4: x_dbl = xproj[k] @ xs,  dt projection + softplus -> delta, split B/C
// 4 blocks per (b,k); each thread owns one l.
// ---------------------------------------------------------------------------
__global__ void k4_xdbl(const float* __restrict__ xs, const float* __restrict__ xproj,
                        const float* __restrict__ dtw, const float* __restrict__ dtb,
                        float* __restrict__ delta, float* __restrict__ Bsw,
                        float* __restrict__ Csw) {
  const int C34 = RLOW + 2 * NST;  // 34
  __shared__ float swp[C34 * DSV];
  __shared__ float sdtw[DSV * RLOW];
  __shared__ float sdtb[DSV];

  int bk = (int)(blockIdx.x >> 2);
  int k  = bk & 3;
  int l  = ((int)(blockIdx.x & 3) << 8) + (int)threadIdx.x;

  for (int i = threadIdx.x; i < C34 * DSV; i += blockDim.x) swp[i] = xproj[k * C34 * DSV + i];
  if (threadIdx.x < DSV * RLOW) sdtw[threadIdx.x] = dtw[k * DSV * RLOW + threadIdx.x];
  if (threadIdx.x < DSV)        sdtb[threadIdx.x] = dtb[k * DSV + threadIdx.x];
  __syncthreads();

  float acc[C34];
#pragma unroll
  for (int c = 0; c < C34; ++c) acc[c] = 0.0f;

  const float* xsp = xs + (size_t)bk * DSV * LSEQ + l;
  for (int d = 0; d < DSV; ++d) {
    float v = xsp[(size_t)d * LSEQ];
#pragma unroll
    for (int c = 0; c < C34; ++c) acc[c] += v * swp[c * DSV + d];
  }

  float dt0 = acc[0], dt1 = acc[1];
  for (int d = 0; d < DSV; ++d) {
    float dv = dt0 * sdtw[d * 2 + 0] + dt1 * sdtw[d * 2 + 1] + sdtb[d];
    delta[((size_t)bk * DSV + d) * LSEQ + l] = softplusf_(dv);
  }
#pragma unroll
  for (int n = 0; n < NST; ++n) {
    Bsw[((size_t)bk * NST + n) * LSEQ + l] = acc[RLOW + n];
    Csw[((size_t)bk * NST + n) * LSEQ + l] = acc[RLOW + NST + n];
  }
}

// ---------------------------------------------------------------------------
// k5: selective scan. One block per (b,k): 512 threads = (d,n) state lattice.
// Chunks of 64 steps staged into double-buffered LDS by the Tensor Data Mover
// (tensor_load_to_lds, TENSORcnt), overlapping DMA of chunk c+1 with the
// recurrence of chunk c. 16-lane shfl_xor reduction over the state dim.
// ---------------------------------------------------------------------------
__global__ void k5_scan(const float* __restrict__ xs, const float* __restrict__ delta,
                        const float* __restrict__ Bsw, const float* __restrict__ Csw,
                        const float* __restrict__ A_logs, const float* __restrict__ Ds,
                        float* __restrict__ yout) {
  const int CH = 64;
  __shared__ float sD[2][DSV * CH];
  __shared__ float sU[2][DSV * CH];
  __shared__ float sB[2][NST * CH];
  __shared__ float sC[2][NST * CH];

  int bk = blockIdx.x;
  int k  = bk & 3;
  int t  = threadIdx.x;
  int d  = t >> 4;
  int n  = t & 15;

  float Acoef = -__expf(A_logs[(k * DSV + d) * NST + n]);
  float Dd    = Ds[k * DSV + d];

  const float* dbase = delta + (size_t)bk * DSV * LSEQ;
  const float* ubase = xs    + (size_t)bk * DSV * LSEQ;
  const float* bbase = Bsw   + (size_t)bk * NST * LSEQ;
  const float* cbase = Csw   + (size_t)bk * NST * LSEQ;
  float*       ybase = yout  + (size_t)bk * DSV * LSEQ;

  bool issuer = (t < 32);  // wave 0 drives the TDM
  if (issuer) {
    tdm_load_2d(dbase, &sD[0][0], CH, DSV, LSEQ, DSV, LSEQ);
    tdm_load_2d(ubase, &sU[0][0], CH, DSV, LSEQ, DSV, LSEQ);
    tdm_load_2d(bbase, &sB[0][0], CH, NST, LSEQ, NST, LSEQ);
    tdm_load_2d(cbase, &sC[0][0], CH, NST, LSEQ, NST, LSEQ);
    __builtin_amdgcn_s_wait_tensorcnt((short)0);
  }
  __syncthreads();

  float h = 0.0f;
  for (int l0 = 0; l0 < LSEQ; l0 += CH) {
    int cur = (l0 >> 6) & 1;
    int nxt = cur ^ 1;

    if ((l0 + CH < LSEQ) && issuer) {   // prefetch next chunk via TDM (async)
      tdm_load_2d(dbase + l0 + CH, &sD[nxt][0], CH, DSV, LSEQ, DSV, LSEQ);
      tdm_load_2d(ubase + l0 + CH, &sU[nxt][0], CH, DSV, LSEQ, DSV, LSEQ);
      tdm_load_2d(bbase + l0 + CH, &sB[nxt][0], CH, NST, LSEQ, NST, LSEQ);
      tdm_load_2d(cbase + l0 + CH, &sC[nxt][0], CH, NST, LSEQ, NST, LSEQ);
    }

    for (int ll = 0; ll < CH; ++ll) {
      float dt = sD[cur][d * CH + ll];
      float u  = sU[cur][d * CH + ll];
      float Bn = sB[cur][n * CH + ll];
      float Cn = sC[cur][n * CH + ll];
      h = __expf(dt * Acoef) * h + dt * u * Bn;
      float yv = h * Cn;
      yv += __shfl_xor(yv, 1, 32);
      yv += __shfl_xor(yv, 2, 32);
      yv += __shfl_xor(yv, 4, 32);
      yv += __shfl_xor(yv, 8, 32);
      if (n == 0) ybase[(size_t)d * LSEQ + l0 + ll] = yv + u * Dd;
    }

    if (issuer) __builtin_amdgcn_s_wait_tensorcnt((short)0);
    __syncthreads();
  }
}

// ---------------------------------------------------------------------------
// k6: inverse cross-merge + SE gating + LayerNorm + z gating
// One block per output row (b,hh,p); 128 threads over channels.
// ---------------------------------------------------------------------------
__global__ void k6_post(const float* __restrict__ yin, const float* __restrict__ f2g,
                        const float* __restrict__ zs, const float* __restrict__ ln_g,
                        const float* __restrict__ ln_b, float* __restrict__ yyn) {
  int row = blockIdx.x;                  // (b*32+hh)*32 + p
  int p  = row & 31;
  int hh = (row >> 5) & 31;
  int b  = row >> 10;
  int c  = threadIdx.x;
  int q  = c >> 2, g = c & 3;

  int l;
  if (g == 0)      l = p * 32 + q;
  else if (g == 1) l = q * 32 + p;
  else if (g == 2) l = 1023 - (p * 32 + q);
  else             l = 1023 - (q * 32 + p);

  float v = yin[(((size_t)b * KDIR + g) * DSV + hh) * LSEQ + l];
  v *= f2g[b * CDIM + c];

  __shared__ float s1[CDIM];
  __shared__ float s2[CDIM];
  s1[c] = v;
  s2[c] = v * v;
  __syncthreads();
  for (int s = 64; s > 0; s >>= 1) {
    if (c < s) { s1[c] += s1[c + s]; s2[c] += s2[c + s]; }
    __syncthreads();
  }
  float mu  = s1[0] * (1.0f / CDIM);
  float var = s2[0] * (1.0f / CDIM) - mu * mu;
  float out = (v - mu) * rsqrtf(var + 1e-5f) * ln_g[c] + ln_b[c];
  out *= zs[(size_t)row * CDIM + c];
  yyn[(size_t)row * CDIM + c] = out;
}

// ---------------------------------------------------------------------------
extern "C" void kernel_launch(void* const* d_in, const int* in_sizes, int n_in,
                              void* d_out, int out_size, void* d_ws, size_t ws_size,
                              hipStream_t stream) {
  const float* x            = (const float*)d_in[0];
  const float* in_proj_w    = (const float*)d_in[1];   // (256,128)
  const float* conv_w       = (const float*)d_in[2];
  const float* conv_b       = (const float*)d_in[3];
  const float* fc1_w        = (const float*)d_in[4];
  const float* fc1_b        = (const float*)d_in[5];
  const float* fc2_w        = (const float*)d_in[6];
  const float* fc2_b        = (const float*)d_in[7];
  const float* x_proj_w     = (const float*)d_in[8];   // (4,34,32)
  const float* dt_projs_w   = (const float*)d_in[9];   // (4,32,2)
  const float* dt_projs_b   = (const float*)d_in[10];  // (4,32)
  const float* A_logs       = (const float*)d_in[11];  // (128,16)
  const float* Ds           = (const float*)d_in[12];  // (128,)
  const float* ln_g         = (const float*)d_in[13];
  const float* ln_b         = (const float*)d_in[14];
  const float* out_proj_w   = (const float*)d_in[15];  // (128,128)

  char* wp = (char*)d_ws;
  auto carve = [&](size_t bytes) -> float* {
    float* p = (float*)wp;
    wp += (bytes + 255) & ~size_t(255);
    return p;
  };
  float* xz    = carve((size_t)PIX * 256 * 4);          // in_proj output
  float* zs    = carve((size_t)PIX * CDIM * 4);         // silu(z)
  float* xs    = carve((size_t)BB * KDIR * DSV * LSEQ * 4);
  float* delta = carve((size_t)BB * KDIR * DSV * LSEQ * 4);
  float* Bsw   = carve((size_t)BB * KDIR * NST * LSEQ * 4);
  float* Csw   = carve((size_t)BB * KDIR * NST * LSEQ * 4);
  float* yw    = carve((size_t)BB * KDIR * DSV * LSEQ * 4);
  float* zzsum = carve((size_t)BB * CDIM * 4);
  float* f2g   = carve((size_t)BB * CDIM * 4);
  float* yyn   = carve((size_t)PIX * CDIM * 4);

  // 1) xz = x @ in_proj_w.T   (M=32768, N=256, K=128) — bf16 WMMA, 16x64/wave
  {
    int waves = (PIX / 16) * (256 / 64);   // 8192
    wmma_gemm_bt<CDIM, 256><<<waves / 8, 256, 0, stream>>>(x, in_proj_w, xz, PIX);
  }

  // 2) elementwise + cross-scan scatter (+ squeeze sums)
  hipMemsetAsync(zzsum, 0, (size_t)BB * CDIM * 4, stream);
  k2_pre<<<(PIX * CDIM) / 256, 256, 0, stream>>>(xz, conv_w, conv_b, zs, xs, zzsum);

  // 3) squeeze-excite
  k3_se<<<BB, CDIM, 0, stream>>>(zzsum, fc1_w, fc1_b, fc2_w, fc2_b, f2g);

  // 4) x_dbl projection, dt projection, softplus
  k4_xdbl<<<BB * KDIR * 4, 256, 0, stream>>>(xs, x_proj_w, dt_projs_w, dt_projs_b,
                                             delta, Bsw, Csw);

  // 5) selective scan (TDM double-buffered)
  k5_scan<<<BB * KDIR, 512, 0, stream>>>(xs, delta, Bsw, Csw, A_logs, Ds, yw);

  // 6) merge + gate + LayerNorm + z gate
  k6_post<<<PIX, CDIM, 0, stream>>>(yw, f2g, zs, ln_g, ln_b, yyn);

  // 7) out = yyn @ out_proj_w.T  (M=32768, N=128, K=128) — bf16 WMMA
  {
    int waves = (PIX / 16) * (CDIM / 64);  // 4096
    wmma_gemm_bt<CDIM, CDIM><<<waves / 8, 256, 0, stream>>>(yyn, out_proj_w,
                                                            (float*)d_out, PIX);
  }
}
